// MixtralSparseMoeBlock_85435489452376
// MI455X (gfx1250) — matrix operand order repaired and checked by reading, not compiled, not run on hardware
//
#include <hip/hip_runtime.h>
#include <math.h>

// Problem constants (from reference)
#define Bb   2
#define Ss   1024
#define Tt   2048          // B*S tokens
#define Hh   2048
#define Ff   7168
#define Ee   8
#define TOPK 2

// Tiling
#define TM   16            // tokens per block tile
#define FC   128           // F chunk width (gate) — also up chunk width
#define KS   32            // WMMA K step (bf16)
#define NQ   512           // output-N quarter in GEMM2

typedef __attribute__((ext_vector_type(16))) __bf16 v16bf;
typedef __attribute__((ext_vector_type(8)))  float  v8f;
typedef __attribute__((ext_vector_type(2)))  __bf16 bf16x2;

union Frag { v16bf v; __bf16 e[16]; };

// ---------------------------------------------------------------------------
// Kernel 1: router — logits, softmax, top-2, renormalized weights
// ---------------------------------------------------------------------------
__global__ __launch_bounds__(256) void router_kernel(
    const float* __restrict__ x, const float* __restrict__ gw,
    float* __restrict__ logits_out, int* __restrict__ sel,
    float* __restrict__ selw) {
  const int t   = blockIdx.x;
  const int tid = threadIdx.x;
  __shared__ float part[256][Ee + 1];
  __shared__ float lg[Ee];

  float p[Ee];
#pragma unroll
  for (int e = 0; e < Ee; ++e) p[e] = 0.f;

  const float* xr = x + (size_t)t * Hh;
  for (int h = tid; h < Hh; h += 256) {
    float xv = xr[h];
    const float* g = gw + (size_t)h * Ee;
#pragma unroll
    for (int e = 0; e < Ee; ++e) p[e] += xv * g[e];
  }
#pragma unroll
  for (int e = 0; e < Ee; ++e) part[tid][e] = p[e];
  __syncthreads();

  if (tid < Ee) {
    float s = 0.f;
    for (int i = 0; i < 256; ++i) s += part[i][tid];
    lg[tid] = s;
    logits_out[(size_t)t * Ee + tid] = s;
  }
  __syncthreads();

  if (tid == 0) {
    float m = lg[0];
#pragma unroll
    for (int e = 1; e < Ee; ++e) m = fmaxf(m, lg[e]);
    float pr[Ee], se = 0.f;
#pragma unroll
    for (int e = 0; e < Ee; ++e) { pr[e] = __expf(lg[e] - m); se += pr[e]; }
    float inv = 1.f / se;
#pragma unroll
    for (int e = 0; e < Ee; ++e) pr[e] *= inv;

    int i0 = 0;
#pragma unroll
    for (int e = 1; e < Ee; ++e) if (pr[e] > pr[i0]) i0 = e;
    int i1 = (i0 == 0) ? 1 : 0;
#pragma unroll
    for (int e = 0; e < Ee; ++e) if (e != i0 && pr[e] > pr[i1]) i1 = e;

    float ws = pr[i0] + pr[i1];
    sel[t * 2 + 0]  = i0;  selw[t * 2 + 0] = pr[i0] / ws;
    sel[t * 2 + 1]  = i1;  selw[t * 2 + 1] = pr[i1] / ws;
  }
}

// ---------------------------------------------------------------------------
// Kernel 2: deterministic per-expert compaction (no atomics)
// ---------------------------------------------------------------------------
__global__ void build_lists(const int* __restrict__ sel,
                            const float* __restrict__ selw,
                            int* __restrict__ counts,
                            int* __restrict__ idxl,
                            float* __restrict__ wl) {
  int e = threadIdx.x;
  if (e >= Ee) return;
  int c = 0;
  for (int t = 0; t < Tt; ++t) {
#pragma unroll
    for (int k = 0; k < TOPK; ++k) {
      if (sel[t * 2 + k] == e) {
        idxl[e * Tt + c] = t;
        wl[e * Tt + c]   = selw[t * 2 + k];
        ++c;
      }
    }
  }
  counts[e] = c;
}

// ---------------------------------------------------------------------------
// Kernel 3: fused expert FFN. grid=(T/TM, E), 256 thr = 8 waves.
// GEMM1: g = silu(x@w1_gate)*(x@w1_up), GEMM2: acc += g @ w2, both bf16 WMMA.
// ---------------------------------------------------------------------------
__global__ __launch_bounds__(256) void moe_kernel(
    const float* __restrict__ x, const float* __restrict__ w1,
    const float* __restrict__ w2, const int* __restrict__ counts,
    const int* __restrict__ idxl, const float* __restrict__ wl,
    float* __restrict__ out) {
  const int e     = blockIdx.y;
  const int tile0 = blockIdx.x * TM;
  const int cnt   = counts[e];
  if (tile0 >= cnt) return;

  const int tid  = threadIdx.x;
  const int wave = tid >> 5;
  const int lane = tid & 31;
  const int lrow = lane & 15;   // N (or M) index within 16
  const int lhi  = lane >> 4;   // high half-wave selector

  // LDS staging (rows padded to 16B multiples so frag reads become ds_load_b128)
  __shared__ __bf16 Abf[TM][Hh + 8];   // x tile, bf16          (65,792 B)
  __shared__ __bf16 Bt1[2 * FC][40];   // w1 slab, [col][k]     (20,480 B)
  __shared__ __bf16 Gbf[TM][FC + 8];   // g chunk, [M][k]       ( 4,352 B)
  __shared__ __bf16 Bt2[NQ][40];       // w2 slab, [col][k]     (40,960 B)
  __shared__ int    toks[TM];
  __shared__ float  tw[TM];

  if (tid < TM) {
    int r = tile0 + tid;
    if (r < cnt) { toks[tid] = idxl[e * Tt + r]; tw[tid] = wl[e * Tt + r]; }
    else         { toks[tid] = 0;               tw[tid] = 0.f; }
  }
  __syncthreads();

  // Stage A: 16 x 2048 f32 -> bf16 (coalesced; one bf16 per thread per iteration)
  for (int i = tid; i < TM * Hh; i += 256) {
    int r = i >> 11, c = i & (Hh - 1);
    Abf[r][c] = (__bf16)x[(size_t)toks[r] * Hh + c];
  }
  __syncthreads();

  v8f acc[16];
#pragma unroll
  for (int i = 0; i < 16; ++i) {
    v8f z = {0.f, 0.f, 0.f, 0.f, 0.f, 0.f, 0.f, 0.f};
    acc[i] = z;
  }

  const size_t w1base = (size_t)e * Hh * (2 * (size_t)Ff);
  const size_t w2base = (size_t)e * Ff * Hh;

  for (int fc = 0; fc < Ff; fc += FC) {
    // ---------------- GEMM1: K = H in steps of 32 ----------------
    v8f cg, cu;
    {
      v8f z = {0.f, 0.f, 0.f, 0.f, 0.f, 0.f, 0.f, 0.f};
      cg = z; cu = z;
    }
    // prefetch next chunk's first w1 row into L2
    if (fc + FC < Ff)
      __builtin_prefetch(w1 + w1base + (fc + FC) + tid, 0, 1);

    for (int k0 = 0; k0 < Hh; k0 += KS) {
      // stage w1 slab transposed: 32 k-rows x 256 cols (gate | up)
      // pack 2 consecutive k per store -> v_cvt_pk_bf16_f32 + ds_store_b32
      {
        int col = (tid < FC) ? (fc + tid) : (Ff + fc + (tid - FC));
        const float* src = w1 + w1base + (size_t)k0 * (2 * Ff) + col;
        bf16x2* dst = (bf16x2*)&Bt1[tid][0];
#pragma unroll 4
        for (int r = 0; r < KS; r += 2) {
          bf16x2 v = {(__bf16)src[(size_t)r * (2 * Ff)],
                      (__bf16)src[(size_t)(r + 1) * (2 * Ff)]};
          dst[r >> 1] = v;
        }
      }
      __syncthreads();

      Frag a, bg, bu;
      {
        const int kb = k0 + lhi * 16;
#pragma unroll
        for (int j = 0; j < 8; ++j) a.e[j]     = Abf[lrow][kb + j];
#pragma unroll
        for (int j = 0; j < 8; ++j) a.e[8 + j] = Abf[lrow][kb + 8 + j];
      }
      {
        const __bf16* bp = &Bt1[wave * 16 + lrow][lhi * 16];
#pragma unroll
        for (int j = 0; j < 16; ++j) bg.e[j] = bp[j];
        bp = &Bt1[FC + wave * 16 + lrow][lhi * 16];
#pragma unroll
        for (int j = 0; j < 16; ++j) bu.e[j] = bp[j];
      }
      cg = __builtin_amdgcn_wmma_f32_16x16x32_bf16(false, a.v, false, bg.v,
                                                   (short)0, cg, false, false);
      cu = __builtin_amdgcn_wmma_f32_16x16x32_bf16(false, a.v, false, bu.v,
                                                   (short)0, cu, false, false);
      __syncthreads();
    }

    // silu(gate) * up  -> Gbf (bf16), chunk-local layout [M][k]
    // fast rcp: one v_rcp_f32 instead of the IEEE div sequence
#pragma unroll
    for (int i = 0; i < 8; ++i) {
      float ga = cg[i];
      float s  = ga * __builtin_amdgcn_rcpf(1.f + __expf(-ga));
      Gbf[i + lhi * 8][wave * 16 + lrow] = (__bf16)(s * cu[i]);
    }
    __syncthreads();

    // ---------------- GEMM2: acc[16x2048] += g[16x128] @ w2 ----------------
    for (int ks = 0; ks < FC; ks += KS) {
      Frag ga2;
      {
        const int kb = ks + lhi * 16;
#pragma unroll
        for (int j = 0; j < 8; ++j) ga2.e[j]     = Gbf[lrow][kb + j];
#pragma unroll
        for (int j = 0; j < 8; ++j) ga2.e[8 + j] = Gbf[lrow][kb + 8 + j];
      }
      for (int q = 0; q < 4; ++q) {
        // stage w2 slab transposed: 32 k-rows x 512 cols, packed b32 stores
        {
          const float* s0 = w2 + w2base + (size_t)(fc + ks) * Hh + q * NQ;
#pragma unroll
          for (int cc = 0; cc < 2; ++cc) {
            int col = tid + cc * 256;
            const float* sp = s0 + col;
            bf16x2* dst = (bf16x2*)&Bt2[col][0];
#pragma unroll 4
            for (int r = 0; r < KS; r += 2) {
              bf16x2 v = {(__bf16)sp[(size_t)r * Hh],
                          (__bf16)sp[(size_t)(r + 1) * Hh]};
              dst[r >> 1] = v;
            }
          }
        }
        __syncthreads();
#pragma unroll
        for (int nt = 0; nt < 4; ++nt) {
          Frag b2;
          const __bf16* bp = &Bt2[wave * 64 + nt * 16 + lrow][lhi * 16];
#pragma unroll
          for (int j = 0; j < 16; ++j) b2.e[j] = bp[j];
          acc[q * 4 + nt] = __builtin_amdgcn_wmma_f32_16x16x32_bf16(
              false, ga2.v, false, b2.v, (short)0, acc[q * 4 + nt], false, false);
        }
        __syncthreads();
      }
    }
  }

  // Epilogue: scale by routing weight, accumulate into out.
  // Each token receives exactly TOPK=2 contributions; 2-operand f32 atomic
  // add is commutative -> deterministic.
#pragma unroll
  for (int q = 0; q < 4; ++q) {
#pragma unroll
    for (int nt = 0; nt < 4; ++nt) {
      v8f c = acc[q * 4 + nt];
      int col = q * NQ + wave * 64 + nt * 16 + lrow;
#pragma unroll
      for (int i = 0; i < 8; ++i) {
        int M = i + lhi * 8;
        float v = c[i] * tw[M];
        unsafeAtomicAdd(&out[(size_t)toks[M] * Hh + col], v);
      }
    }
  }
}

// ---------------------------------------------------------------------------
extern "C" void kernel_launch(void* const* d_in, const int* in_sizes, int n_in,
                              void* d_out, int out_size, void* d_ws,
                              size_t ws_size, hipStream_t stream) {
  const float* x  = (const float*)d_in[0];   // [T, H]
  const float* gw = (const float*)d_in[1];   // [H, E]
  const float* w1 = (const float*)d_in[2];   // [E, H, 2F]
  const float* w2 = (const float*)d_in[3];   // [E, F, H]

  float* out    = (float*)d_out;                       // [T, H]
  float* logits = out + (size_t)Tt * Hh;               // [T, E]

  // workspace layout
  char*  ws     = (char*)d_ws;
  int*   sel    = (int*)(ws);                          // T*2 ints
  float* selw   = (float*)(ws + Tt * 2 * 4);           // T*2 floats
  int*   counts = (int*)(ws + Tt * 4 * 4);             // E ints (padded)
  int*   idxl   = (int*)(ws + Tt * 4 * 4 + 64);        // E*T ints
  float* wl     = (float*)(ws + Tt * 4 * 4 + 64 + Ee * Tt * 4);

  hipMemsetAsync(out, 0, (size_t)Tt * Hh * sizeof(float), stream);
  router_kernel<<<Tt, 256, 0, stream>>>(x, gw, logits, sel, selw);
  build_lists<<<1, 32, 0, stream>>>(sel, selw, counts, idxl, wl);
  moe_kernel<<<dim3(Tt / TM, Ee), 256, 0, stream>>>(x, w1, w2, counts, idxl,
                                                    wl, out);
}
